// IGAB_22239340658786
// MI455X (gfx1250) — matrix operand order repaired and checked by reading, not compiled, not run on hardware
//
#include <hip/hip_runtime.h>
#include <hip/hip_fp16.h>

#define B_    2
#define C_    64
#define H_    256
#define W_    256
#define HEADS_ 8
#define HID_  256
#define HWsz  (H_ * W_)            // 65536
#define NTOK  (B_ * HWsz)          // 131072

typedef __attribute__((ext_vector_type(16))) _Float16 v16h;
typedef __attribute__((ext_vector_type(8)))  _Float16 v8h;
typedef __attribute__((ext_vector_type(8)))  float    v8f;

// ---------------------------------------------------------------- helpers

__device__ __forceinline__ v8f wmma_f16(v16h a, v16h b, v8f c) {
  return __builtin_amdgcn_wmma_f32_16x16x32_f16(
      /*neg_a=*/false, a, /*neg_b=*/false, b,
      /*c_mod=*/(short)0, c, /*reuse_a=*/false, /*reuse_b=*/false);
}

__device__ __forceinline__ float gelu_exact(float x) {
  return 0.5f * x * (1.0f + erff(x * 0.70710678118654752f));
}

__device__ __forceinline__ float wave_red32(float v) {
#pragma unroll
  for (int off = 16; off > 0; off >>= 1) v += __shfl_xor(v, off, 32);
  return v;
}

// A fragment (16 x 32, f16 row-major, leading dim ld): two contiguous b128 loads.
// lane layout: M = lane&15 ; element j -> K = kBase + 8*(lane>>4) + (j&7) + 16*(j>>3)
__device__ __forceinline__ v16h load_a_h(const _Float16* A, int ld, int kBase, int lane) {
  const _Float16* p = A + (size_t)(lane & 15) * ld + kBase + ((lane >> 4) << 3);
  v8h lo = *(const v8h*)p;
  v8h hi = *(const v8h*)(p + 16);
  return __builtin_shufflevector(lo, hi, 0, 1, 2, 3, 4, 5, 6, 7,
                                 8, 9, 10, 11, 12, 13, 14, 15);
}

// B fragment from pre-packed fragment-order buffer: 32B contiguous per lane.
__device__ __forceinline__ v16h load_b_pk(const _Float16* P, int frag, int lane) {
  return *(const v16h*)(P + (((size_t)frag * 32 + lane) << 4));
}

// ---------------------------------------------------------------- layout kernels

// NCHW -> NHWC producing both f32 and f16 copies
__global__ void nchw2nhwc2_k(const float* __restrict__ src, float* __restrict__ dst,
                             _Float16* __restrict__ dsth) {
  int idx = blockIdx.x * blockDim.x + threadIdx.x;      // NCHW linear
  int hw = idx % HWsz;
  int c  = (idx / HWsz) % C_;
  int b  = idx / (HWsz * C_);
  float v = src[idx];
  size_t o = ((size_t)b * HWsz + hw) * C_ + c;
  dst[o]  = v;
  dsth[o] = (_Float16)v;
}

__global__ void nchw2nhwc_k(const float* __restrict__ src, float* __restrict__ dst) {
  int idx = blockIdx.x * blockDim.x + threadIdx.x;
  int hw = idx % HWsz;
  int c  = (idx / HWsz) % C_;
  int b  = idx / (HWsz * C_);
  dst[((size_t)b * HWsz + hw) * C_ + c] = src[idx];
}

__global__ void final_out_k(const float* __restrict__ x1, const float* __restrict__ x,
                            float* __restrict__ out) {
  int idx = blockIdx.x * blockDim.x + threadIdx.x;      // NCHW linear
  int hw = idx % HWsz;
  int c  = (idx / HWsz) % C_;
  int b  = idx / (HWsz * C_);
  out[idx] = x1[((size_t)b * HWsz + hw) * C_ + c] + x[idx];
}

__global__ void zero_k(float* p, int n) {
  int i = blockIdx.x * blockDim.x + threadIdx.x;
  if (i < n) p[i] = 0.0f;
}

// Pack fp32 weights [K, Nout] into f16 WMMA-B fragment order:
// out[((ct*(K/32)+kh)*32 + lane)*16 + j] = W[(kh*32 + 8*(lane>>4) + (j&7) + 16*(j>>3))*Nout + ct*16 + (lane&15)]
__global__ void pack_b_k(const float* __restrict__ W, _Float16* __restrict__ out,
                         int K, int Nout) {
  int idx = blockIdx.x * blockDim.x + threadIdx.x;
  if (idx >= K * Nout) return;
  int j    = idx & 15;
  int lane = (idx >> 4) & 31;
  int rest = idx >> 9;
  int KH = K >> 5;
  int kh = rest % KH;
  int ct = rest / KH;
  int kk  = kh * 32 + ((lane >> 4) << 3) + (j & 7) + ((j >> 3) << 4);
  int col = ct * 16 + (lane & 15);
  out[idx] = (_Float16)W[kk * Nout + col];
}

// ---------------------------------------------------------------- GEMM kernels (WMMA)

// q/k/v projections fused; vgh = f16(v * illu). A: x1h f16 [N,64]; B: packed 3x4096 halfs.
__global__ __launch_bounds__(256)
void gemm_qkv_k(const _Float16* __restrict__ x1h, const float* __restrict__ illu,
                const _Float16* __restrict__ Wp3,
                float* __restrict__ q, float* __restrict__ k,
                float* __restrict__ v, _Float16* __restrict__ vgh) {
  int wave = (blockIdx.x * blockDim.x + threadIdx.x) >> 5;
  int lane = threadIdx.x & 31;
  int mBase = wave * 16;
  const _Float16* A = x1h + (size_t)mBase * C_;
  v16h a0 = load_a_h(A, C_, 0, lane);
  v16h a1 = load_a_h(A, C_, 32, lane);
  float* Os[3] = {q, k, v};
#pragma unroll
  for (int p = 0; p < 3; ++p) {
    const _Float16* P = Wp3 + p * 4096;
#pragma unroll
    for (int ct = 0; ct < 4; ++ct) {
      v8f c = {0, 0, 0, 0, 0, 0, 0, 0};
      c = wmma_f16(a0, load_b_pk(P, ct * 2 + 0, lane), c);
      c = wmma_f16(a1, load_b_pk(P, ct * 2 + 1, lane), c);
      int col = ct * 16 + (lane & 15);
      int mo  = (lane >> 4) * 8;
#pragma unroll
      for (int r = 0; r < 8; ++r) {
        size_t o = (size_t)(mBase + mo + r) * C_ + col;
        float val = c[r];
        Os[p][o] = val;
        if (p == 2) vgh[o] = (_Float16)(val * illu[o]);
      }
    }
  }
}

// attention apply + output projection collapsed: out = vg @ W_eff[b] + bp
__global__ __launch_bounds__(256)
void gemm_apply_k(const _Float16* __restrict__ vgh, const _Float16* __restrict__ weffp,
                  const float* __restrict__ bp, float* __restrict__ out) {
  int wave = (blockIdx.x * blockDim.x + threadIdx.x) >> 5;
  int lane = threadIdx.x & 31;
  int mBase = wave * 16;
  const _Float16* P = weffp + (size_t)(mBase / HWsz) * 4096;
  const _Float16* A = vgh + (size_t)mBase * C_;
  v16h a0 = load_a_h(A, C_, 0, lane);
  v16h a1 = load_a_h(A, C_, 32, lane);
#pragma unroll
  for (int ct = 0; ct < 4; ++ct) {
    v8f c = {0, 0, 0, 0, 0, 0, 0, 0};
    c = wmma_f16(a0, load_b_pk(P, ct * 2 + 0, lane), c);
    c = wmma_f16(a1, load_b_pk(P, ct * 2 + 1, lane), c);
    int col = ct * 16 + (lane & 15);
    int mo  = (lane >> 4) * 8;
#pragma unroll
    for (int r = 0; r < 8; ++r)
      out[(size_t)(mBase + mo + r) * C_ + col] = c[r] + bp[col];
  }
}

// FFN stage 1: y1 = gelu(lnh @ w1)  [N,64]x[64,256] -> f16
__global__ __launch_bounds__(256)
void gemm_ff1_k(const _Float16* __restrict__ lnh, const _Float16* __restrict__ w1p,
                _Float16* __restrict__ y1) {
  int wave = (blockIdx.x * blockDim.x + threadIdx.x) >> 5;
  int lane = threadIdx.x & 31;
  int mBase = wave * 16;
  const _Float16* A = lnh + (size_t)mBase * C_;
  v16h a0 = load_a_h(A, C_, 0, lane);
  v16h a1 = load_a_h(A, C_, 32, lane);
#pragma unroll
  for (int ct = 0; ct < 16; ++ct) {
    v8f c = {0, 0, 0, 0, 0, 0, 0, 0};
    c = wmma_f16(a0, load_b_pk(w1p, ct * 2 + 0, lane), c);
    c = wmma_f16(a1, load_b_pk(w1p, ct * 2 + 1, lane), c);
    int col = ct * 16 + (lane & 15);
    int mo  = (lane >> 4) * 8;
#pragma unroll
    for (int r = 0; r < 8; ++r)
      y1[(size_t)(mBase + mo + r) * HID_ + col] = (_Float16)gelu_exact(c[r]);
  }
}

// FFN stage 2: x1 += y2 @ w2 ; also refresh x1h shadow
__global__ __launch_bounds__(256)
void gemm_ff2_k(const _Float16* __restrict__ y2, const _Float16* __restrict__ w2p,
                float* __restrict__ x1, _Float16* __restrict__ x1h) {
  int wave = (blockIdx.x * blockDim.x + threadIdx.x) >> 5;
  int lane = threadIdx.x & 31;
  int mBase = wave * 16;
  const _Float16* A = y2 + (size_t)mBase * HID_;
  v16h af[8];
#pragma unroll
  for (int kh = 0; kh < 8; ++kh) af[kh] = load_a_h(A, HID_, kh * 32, lane);
#pragma unroll
  for (int ct = 0; ct < 4; ++ct) {
    v8f c = {0, 0, 0, 0, 0, 0, 0, 0};
#pragma unroll
    for (int kh = 0; kh < 8; ++kh)
      c = wmma_f16(af[kh], load_b_pk(w2p, ct * 8 + kh, lane), c);
    int col = ct * 16 + (lane & 15);
    int mo  = (lane >> 4) * 8;
#pragma unroll
    for (int r = 0; r < 8; ++r) {
      size_t idx = (size_t)(mBase + mo + r) * C_ + col;
      float nv = x1[idx] + c[r];
      x1[idx]  = nv;
      x1h[idx] = (_Float16)nv;
    }
  }
}

// ---------------------------------------------------------------- attention reductions

__global__ __launch_bounds__(256)
void gram_k(const float* __restrict__ q, const float* __restrict__ k,
            float* __restrict__ G, float* __restrict__ qn, float* __restrict__ kn) {
  int blk   = blockIdx.x;            // B*HEADS*64 blocks
  int chunk = blk & 63;
  int h     = (blk >> 6) & 7;
  int b     = blk >> 9;
  int lane  = threadIdx.x & 31;
  float g[64], qs[8], ks[8];
#pragma unroll
  for (int t = 0; t < 64; ++t) g[t] = 0.0f;
#pragma unroll
  for (int t = 0; t < 8; ++t) { qs[t] = 0.0f; ks[t] = 0.0f; }
  for (int it = 0; it < 4; ++it) {
    size_t n = (size_t)b * HWsz + chunk * 1024 + it * 256 + threadIdx.x;
    const float* qp = q + n * C_ + h * 8;
    const float* kp = k + n * C_ + h * 8;
    float qv[8], kv[8];
#pragma unroll
    for (int d = 0; d < 8; ++d) {
      qv[d] = qp[d]; kv[d] = kp[d];
      qs[d] += qv[d] * qv[d]; ks[d] += kv[d] * kv[d];
    }
#pragma unroll
    for (int d = 0; d < 8; ++d)
#pragma unroll
      for (int e = 0; e < 8; ++e) g[d * 8 + e] += kv[d] * qv[e];
  }
#pragma unroll
  for (int t = 0; t < 64; ++t) {
    float r = wave_red32(g[t]);
    if (lane == 0) atomicAdd(&G[(b * 8 + h) * 64 + t], r);
  }
#pragma unroll
  for (int d = 0; d < 8; ++d) {
    float r0 = wave_red32(qs[d]);
    float r1 = wave_red32(ks[d]);
    if (lane == 0) {
      atomicAdd(&qn[b * C_ + h * 8 + d], r0);
      atomicAdd(&kn[b * C_ + h * 8 + d], r1);
    }
  }
}

__global__ void attn_k(const float* __restrict__ G, const float* __restrict__ qn,
                       const float* __restrict__ kn, const float* __restrict__ rescale,
                       float* __restrict__ attn) {
  int tid = threadIdx.x;
  if (tid >= B_ * HEADS_ * 8) return;
  int b = tid >> 6, h = (tid >> 3) & 7, d = tid & 7;
  float knd = fmaxf(sqrtf(kn[b * C_ + h * 8 + d]), 1e-12f);
  float lg[8], mx = -1e30f;
#pragma unroll
  for (int e = 0; e < 8; ++e) {
    float qne = fmaxf(sqrtf(qn[b * C_ + h * 8 + e]), 1e-12f);
    lg[e] = G[(b * 8 + h) * 64 + d * 8 + e] / (knd * qne) * rescale[h];
    mx = fmaxf(mx, lg[e]);
  }
  float s = 0.0f;
#pragma unroll
  for (int e = 0; e < 8; ++e) { lg[e] = expf(lg[e] - mx); s += lg[e]; }
  float inv = 1.0f / s;
#pragma unroll
  for (int e = 0; e < 8; ++e) attn[(b * 8 + h) * 64 + d * 8 + e] = lg[e] * inv;
}

// W_eff[b] = BlockDiag(attn[b]^T) @ Wp  (per-batch 64x64, fp32)
__global__ void weff_k(const float* __restrict__ attn, const float* __restrict__ Wp,
                       float* __restrict__ weff) {
  int idx = blockIdx.x * blockDim.x + threadIdx.x;     // B*64*64
  if (idx >= B_ * C_ * C_) return;
  int b = idx >> 12, cin = (idx >> 6) & 63, cout = idx & 63;
  int h = cin >> 3, e = cin & 7;
  float s = 0.0f;
#pragma unroll
  for (int d = 0; d < 8; ++d)
    s += attn[(b * 8 + h) * 64 + d * 8 + e] * Wp[(h * 8 + d) * C_ + cout];
  weff[idx] = s;
}

// ---------------------------------------------------------------- depthwise convs / LN

__global__ __launch_bounds__(256)
void dwconv64_gelu_k(const float* __restrict__ in, const float* __restrict__ w,
                     float* __restrict__ out) {
  int idx = blockIdx.x * blockDim.x + threadIdx.x;     // N * 64
  int c = idx & 63;
  int pix = idx >> 6;
  int b = pix / HWsz, hw = pix - b * HWsz;
  int y = hw >> 8, x = hw & 255;
  float s = 0.0f;
#pragma unroll
  for (int ky = 0; ky < 3; ++ky) {
    int yy = y + ky - 1; if (yy < 0 || yy >= H_) continue;
#pragma unroll
    for (int kx = 0; kx < 3; ++kx) {
      int xx = x + kx - 1; if (xx < 0 || xx >= W_) continue;
      s += in[(((size_t)b * H_ + yy) * W_ + xx) * C_ + c] * w[(ky * 3 + kx) * C_ + c];
    }
  }
  out[idx] = gelu_exact(s);
}

// x1 += attn_out + dwconv(t, w)   (fused MSA residual)
__global__ __launch_bounds__(256)
void dwconv64_add_k(const float* __restrict__ in, const float* __restrict__ w,
                    const float* __restrict__ attn_out, float* __restrict__ x1) {
  int idx = blockIdx.x * blockDim.x + threadIdx.x;
  int c = idx & 63;
  int pix = idx >> 6;
  int b = pix / HWsz, hw = pix - b * HWsz;
  int y = hw >> 8, x = hw & 255;
  float s = 0.0f;
#pragma unroll
  for (int ky = 0; ky < 3; ++ky) {
    int yy = y + ky - 1; if (yy < 0 || yy >= H_) continue;
#pragma unroll
    for (int kx = 0; kx < 3; ++kx) {
      int xx = x + kx - 1; if (xx < 0 || xx >= W_) continue;
      s += in[(((size_t)b * H_ + yy) * W_ + xx) * C_ + c] * w[(ky * 3 + kx) * C_ + c];
    }
  }
  x1[idx] = x1[idx] + attn_out[idx] + s;
}

// 256-channel depthwise conv (f16 in/out) + gelu, with row-ahead prefetch
__global__ __launch_bounds__(256)
void dwconv256_gelu_k(const _Float16* __restrict__ in, const float* __restrict__ w,
                      _Float16* __restrict__ out) {
  size_t idx = (size_t)blockIdx.x * blockDim.x + threadIdx.x;   // N * 256
  int c = (int)(idx & 255);
  int pix = (int)(idx >> 8);
  int b = pix / HWsz, hw = pix - b * HWsz;
  int y = hw >> 8, x = hw & 255;
  if (y + 2 < H_)
    __builtin_prefetch(in + (((size_t)b * H_ + y + 2) * W_ + x) * HID_ + c, 0, 3);
  float s = 0.0f;
#pragma unroll
  for (int ky = 0; ky < 3; ++ky) {
    int yy = y + ky - 1; if (yy < 0 || yy >= H_) continue;
#pragma unroll
    for (int kx = 0; kx < 3; ++kx) {
      int xx = x + kx - 1; if (xx < 0 || xx >= W_) continue;
      s += (float)in[(((size_t)b * H_ + yy) * W_ + xx) * HID_ + c] * w[(ky * 3 + kx) * HID_ + c];
    }
  }
  out[idx] = (_Float16)gelu_exact(s);
}

// LayerNorm over 64 channels; one wave per token; f16 output for WMMA A-side
__global__ __launch_bounds__(256)
void ln_k(const float* __restrict__ x, const float* __restrict__ g,
          const float* __restrict__ b, _Float16* __restrict__ y) {
  int wave = (blockIdx.x * blockDim.x + threadIdx.x) >> 5;   // token
  int lane = threadIdx.x & 31;
  size_t base = (size_t)wave * C_;
  float v0 = x[base + lane * 2], v1 = x[base + lane * 2 + 1];
  float mu = wave_red32(v0 + v1) * (1.0f / 64.0f);
  float d0 = v0 - mu, d1 = v1 - mu;
  float var = wave_red32(d0 * d0 + d1 * d1) * (1.0f / 64.0f);
  float inv = rsqrtf(var + 1e-5f);
  y[base + lane * 2]     = (_Float16)(d0 * inv * g[lane * 2]     + b[lane * 2]);
  y[base + lane * 2 + 1] = (_Float16)(d1 * inv * g[lane * 2 + 1] + b[lane * 2 + 1]);
}

// ---------------------------------------------------------------- launcher

extern "C" void kernel_launch(void* const* d_in, const int* in_sizes, int n_in,
                              void* d_out, int out_size, void* d_ws, size_t ws_size,
                              hipStream_t stream) {
  (void)in_sizes; (void)n_in; (void)out_size; (void)ws_size;
  const float* x      = (const float*)d_in[0];
  const float* illu   = (const float*)d_in[1];
  const float* Wq     = (const float*)d_in[2];
  const float* Wk     = (const float*)d_in[3];
  const float* Wv     = (const float*)d_in[4];
  const float* rsc    = (const float*)d_in[5];
  const float* Wp     = (const float*)d_in[6];
  const float* bp     = (const float*)d_in[7];
  const float* pw1    = (const float*)d_in[8];
  const float* pw2    = (const float*)d_in[9];
  const float* ln_g   = (const float*)d_in[10];
  const float* ln_b   = (const float*)d_in[11];
  const float* ff_w1  = (const float*)d_in[12];
  const float* ff_dw  = (const float*)d_in[13];
  const float* ff_w2  = (const float*)d_in[14];
  float* out = (float*)d_out;

  float* ws = (float*)d_ws;
  const size_t NC = (size_t)NTOK * C_;     // 8.39M floats
  float* x1       = ws;                    // [NC] residual stream f32
  float* illuT    = ws + 1 * NC;           // [NC]
  float* qbuf     = ws + 2 * NC;           // q f32 -> dw-tmp f32 -> y1 f16
  float* kbuf     = ws + 3 * NC;           // k f32 -> lnh f16 -> y2 f16
  float* v        = ws + 4 * NC;           // [NC] f32
  float* attn_out = ws + 5 * NC;           // [NC] f32
  _Float16* x1h   = (_Float16*)(ws + 6 * NC);              // NC halfs
  _Float16* vgh   = (_Float16*)(ws + 6 * NC + NC / 2);     // NC halfs
  float* small    = ws + 7 * NC;
  float* G     = small;          // 1024
  float* qn    = G + 1024;       // 128
  float* kn    = qn + 128;       // 128
  float* attnS = kn + 128;       // 1024
  float* weff  = attnS + 1024;   // 8192
  _Float16* wqkv_p = (_Float16*)(weff + 8192);       // 12288 halfs
  _Float16* weff_p = wqkv_p + 3 * 4096;              // 8192 halfs
  _Float16* ff1_p  = weff_p + 2 * 4096;              // 16384 halfs
  _Float16* ff2_p  = ff1_p + 16384;                  // 16384 halfs

  float*    dwt = qbuf;
  _Float16* y1  = (_Float16*)qbuf;
  _Float16* lnh = (_Float16*)kbuf;
  _Float16* y2  = (_Float16*)kbuf;

  const int T = 256;
  const int gElem64 = (int)(NC / T);                     // 32768
  const int gGemm   = (NTOK / 16) / 8;                   // 1024
  const int gDw256  = (int)((size_t)NTOK * HID_ / T);    // 131072

  nchw2nhwc2_k<<<gElem64, T, 0, stream>>>(x, x1, x1h);
  nchw2nhwc_k<<<gElem64, T, 0, stream>>>(illu, illuT);

  for (int i = 0; i < 4; ++i) {
    // ---- weight packing (f16 WMMA fragment order) ----
    pack_b_k<<<16, T, 0, stream>>>(Wq + i * 4096, wqkv_p,          64, 64);
    pack_b_k<<<16, T, 0, stream>>>(Wk + i * 4096, wqkv_p + 4096,   64, 64);
    pack_b_k<<<16, T, 0, stream>>>(Wv + i * 4096, wqkv_p + 8192,   64, 64);
    pack_b_k<<<64, T, 0, stream>>>(ff_w1 + i * (C_ * HID_), ff1_p, 64, 256);
    pack_b_k<<<64, T, 0, stream>>>(ff_w2 + i * (HID_ * C_), ff2_p, 256, 64);

    // ---- MSA ----
    gemm_qkv_k<<<gGemm, T, 0, stream>>>(x1h, illuT, wqkv_p, qbuf, kbuf, v, vgh);
    zero_k<<<5, T, 0, stream>>>(G, 1280);
    gram_k<<<B_ * HEADS_ * 64, T, 0, stream>>>(qbuf, kbuf, G, qn, kn);
    attn_k<<<1, 128, 0, stream>>>(G, qn, kn, rsc + i * HEADS_, attnS);
    weff_k<<<(B_ * C_ * C_) / T, T, 0, stream>>>(attnS, Wp + i * 4096, weff);
    pack_b_k<<<16, T, 0, stream>>>(weff,        weff_p,        64, 64);
    pack_b_k<<<16, T, 0, stream>>>(weff + 4096, weff_p + 4096, 64, 64);
    gemm_apply_k<<<gGemm, T, 0, stream>>>(vgh, weff_p, bp + i * C_, attn_out);
    dwconv64_gelu_k<<<gElem64, T, 0, stream>>>(v, pw1 + i * 576, dwt);
    dwconv64_add_k<<<gElem64, T, 0, stream>>>(dwt, pw2 + i * 576, attn_out, x1);

    // ---- FFN ----
    ln_k<<<NTOK / 8, T, 0, stream>>>(x1, ln_g + i * C_, ln_b + i * C_, lnh);
    gemm_ff1_k<<<gGemm, T, 0, stream>>>(lnh, ff1_p, y1);
    dwconv256_gelu_k<<<gDw256, T, 0, stream>>>(y1, ff_dw + i * (9 * HID_), y2);
    gemm_ff2_k<<<gGemm, T, 0, stream>>>(y2, ff2_p, x1, x1h);
  }

  final_out_k<<<gElem64, T, 0, stream>>>(x1, x, out);
}
